// LSTM_21990232556240
// MI455X (gfx1250) — matrix operand order repaired and checked by reading, not compiled, not run on hardware
//
#include <hip/hip_runtime.h>
#include <stdint.h>

typedef __attribute__((ext_vector_type(16))) __bf16 v16bf;
typedef __attribute__((ext_vector_type(8)))  float  v8f;
typedef unsigned int u32x4 __attribute__((ext_vector_type(4)));

#define S_DIM 1024
#define V_DIM 32000
#define B_DIM 8
#define T_DIM 256
#define ROW16 (16 * S_DIM)   // A-side buffers padded to 16 rows (WMMA M-tile)

// ---------------- device helpers ----------------

__device__ __forceinline__ unsigned short f2bf(float f) {
  union { float f; unsigned u; } x; x.f = f;
  unsigned r = x.u + 0x7FFFu + ((x.u >> 16) & 1u);   // round-to-nearest-even
  return (unsigned short)(r >> 16);
}

__device__ __forceinline__ v8f wmma_bf16(v16bf a, v16bf b, v8f c) {
  // D = A(16x32) * B(32x16) + C, f32 accumulate
  return __builtin_amdgcn_wmma_f32_16x16x32_bf16(false, a, false, b, (short)0, c,
                                                 false, false);
}

// A tile (16x32 bf16). Lane l: row M=l%16, K = {kk+8h..+7} U {kk+16+8h..+7}, h=l/16.
__device__ __forceinline__ v16bf load_a_row(const unsigned short* rowptr, int kk, int half) {
  union { v16bf v; u32x4 q[2]; } u;
  const unsigned short* p = rowptr + kk + half * 8;
  u.q[0] = *(const u32x4*)p;
  u.q[1] = *(const u32x4*)(p + 16);
  return u.v;
}

// B tile (32x16 bf16). Lane l: col N=l%16, K = {kk+8h..+7} U {kk+16+8h..+7}, h=l/16.
__device__ __forceinline__ v16bf load_b_row(const unsigned short* rowptr, int kk, int half) {
  union { v16bf v; u32x4 q[2]; } u;
  const unsigned short* p = rowptr + kk + half * 8;
  u.q[0] = *(const u32x4*)p;
  u.q[1] = *(const u32x4*)(p + 16);
  return u.v;
}

__device__ __forceinline__ float sigmoidf_(float x) { return 1.0f / (1.0f + expf(-x)); }

// ---------------- utility kernels ----------------

__global__ void zero_ws_kernel(unsigned int* __restrict__ p, int nwords) {
  int i = blockIdx.x * blockDim.x + threadIdx.x;
  if (i < nwords) p[i] = 0u;
}

__global__ void convert_bf16_kernel(const float* __restrict__ src,
                                    unsigned short* __restrict__ dst, int n) {
  int i = blockIdx.x * blockDim.x + threadIdx.x;
  if (i < n) dst[i] = f2bf(src[i]);
}

// inp_seq[t][m][s] = bf16(emb[x[m][t]][s]) for m<8, 0 for padding rows m=8..15
__global__ void gather_emb_kernel(const int* __restrict__ x,
                                  const float* __restrict__ emb,
                                  unsigned short* __restrict__ inp_seq) {
  int i = blockIdx.x * blockDim.x + threadIdx.x;   // over T*16*S
  int s = i & (S_DIM - 1);
  int m = (i >> 10) & 15;
  int t = i >> 14;
  unsigned short v = 0;
  if (m < B_DIM) {
    int tok = x[m * T_DIM + t];
    v = f2bf(emb[(size_t)tok * S_DIM + s]);
  }
  inp_seq[i] = v;
}

// ---------------- recurrent step kernel ----------------
// Blocks [0,n1): layer1 at t1. Blocks [n1,...): layer0 at t0.
// 8 waves/WG split K (128 each); each WG owns 16 cell columns (s-block).
__global__ __launch_bounds__(256) void lstm_step_kernel(
    const unsigned short* __restrict__ Wx_bf,   // [2][4096][1024] bf16
    const unsigned short* __restrict__ Wh_bf,
    const unsigned short* __restrict__ inp_seq, // [T][16][1024] bf16 (rows 8..15 zero)
    unsigned short* __restrict__ h0_buf,        // [2][16][1024] bf16 (parity, padded)
    unsigned short* __restrict__ h1_buf,        // [16][1024] bf16 (padded)
    float* __restrict__ h_f32,                  // [2][8][1024]
    float* __restrict__ c_pre,                  // [2][8][1024] pre-norm cell
    float* __restrict__ mu_buf,                 // [2][8]
    float* __restrict__ rstd_buf,               // [2][8]
    float* __restrict__ partials,               // [2][64][8][2]
    unsigned int* __restrict__ counters,        // [2]
    unsigned short* __restrict__ H_all,         // [T*8][1024] bf16
    const float* __restrict__ gamma,            // [2][1024]
    const float* __restrict__ beta,             // [2][1024]
    int t1, int t0, int n1) {
  __shared__ float lds[8 * 4 * 256];   // 32 KB: per-wave partial C tiles
  __shared__ int s_last;

  const int tid  = threadIdx.x;
  const int wave = tid >> 5;
  const int lane = tid & 31;
  const int half = lane >> 4;
  const int l16  = lane & 15;

  int layer, t, sb;
  if ((int)blockIdx.x < n1) { layer = 1; t = t1; sb = blockIdx.x; }
  else                      { layer = 0; t = t0; sb = blockIdx.x - n1; }

  const unsigned short* inpA;
  const unsigned short* hA;
  if (layer == 0) {
    inpA = inp_seq + (size_t)t * ROW16;
    hA   = h0_buf + (size_t)((t - 1) & 1) * ROW16;
  } else {
    inpA = h0_buf + (size_t)(t & 1) * ROW16;
    hA   = h1_buf;
  }
  const size_t gstride = (size_t)S_DIM * S_DIM;           // rows per gate block
  const unsigned short* Wx = Wx_bf + (size_t)layer * 4 * gstride;
  const unsigned short* Wh = Wh_bf + (size_t)layer * 4 * gstride;

  const unsigned short* arow_i = inpA + (size_t)l16 * S_DIM;
  const unsigned short* arow_h = hA   + (size_t)l16 * S_DIM;
  const int nrow = sb * 16 + l16;
  const unsigned short* wxrow = Wx + (size_t)nrow * S_DIM;
  const unsigned short* whrow = Wh + (size_t)nrow * S_DIM;

  v8f vz = {};
  v8f acc[4]; acc[0] = vz; acc[1] = vz; acc[2] = vz; acc[3] = vz;

  #pragma unroll
  for (int kc = 0; kc < 4; ++kc) {
    const int kk = wave * 128 + kc * 32;
    v16bf ax = load_a_row(arow_i, kk, half);   // padding rows are physical zeros
    v16bf ah = load_a_row(arow_h, kk, half);
    #pragma unroll
    for (int g = 0; g < 4; ++g) {
      v16bf bx = load_b_row(wxrow + (size_t)g * gstride, kk, half);
      acc[g] = wmma_bf16(ax, bx, acc[g]);
      v16bf bh = load_b_row(whrow + (size_t)g * gstride, kk, half);
      acc[g] = wmma_bf16(ah, bh, acc[g]);
    }
  }

  // stash partial C tiles: lds[(wave*4+g)*256 + v*32 + lane]
  #pragma unroll
  for (int g = 0; g < 4; ++g) {
    float* dst = lds + (((wave << 2) + g) << 8) + lane;
    #pragma unroll
    for (int v = 0; v < 8; ++v) dst[v * 32] = acc[g][v];
  }
  __syncthreads();

  // Elementwise cell update: tid<128 -> (b = tid/16, so = tid%16)
  float c_new = 0.0f;
  const int b  = tid >> 4;
  const int so = tid & 15;
  const int s  = sb * 16 + so;
  if (tid < 128) {
    float gv[4];
    #pragma unroll
    for (int g = 0; g < 4; ++g) {
      float sum = 0.0f;
      #pragma unroll
      for (int w = 0; w < 8; ++w) sum += lds[(((w << 2) + g) << 8) + b * 32 + so];
      gv[g] = sum;                    // C layout: VGPR r=b (lanes 0-15), N=so
    }
    // lazily normalize the carried cell state from last step's stats
    const float mub = mu_buf[layer * 8 + b];
    const float rsb = rstd_buf[layer * 8 + b];
    const float cp  = c_pre[((size_t)layer * 8 + b) * S_DIM + s];
    const float c_old = (cp - mub) * rsb * gamma[layer * S_DIM + s] + beta[layer * S_DIM + s];

    const float ig = sigmoidf_(gv[0]);
    const float fg = sigmoidf_(gv[1]);
    const float og = sigmoidf_(gv[2]);
    const float gg = tanhf(gv[3]);
    c_new = fg * c_old + ig * gg;
    const float h_new = og * tanhf(c_new);

    c_pre[((size_t)layer * 8 + b) * S_DIM + s] = c_new;
    h_f32[((size_t)layer * 8 + b) * S_DIM + s] = h_new;
    const unsigned short hb = f2bf(h_new);
    if (layer == 0) {
      h0_buf[(size_t)(t & 1) * ROW16 + b * S_DIM + s] = hb;
    } else {
      h1_buf[b * S_DIM + s] = hb;
      H_all[((size_t)t * B_DIM + b) * S_DIM + s] = hb;
    }
  }
  __syncthreads();
  if (tid < 128) { lds[tid] = c_new; lds[128 + tid] = c_new * c_new; }
  __syncthreads();
  if (tid < 8) {   // per-batch partial LN stats over this WG's 16 columns
    float s1 = 0.0f, s2 = 0.0f;
    #pragma unroll
    for (int j = 0; j < 16; ++j) { s1 += lds[tid * 16 + j]; s2 += lds[128 + tid * 16 + j]; }
    partials[(((size_t)layer * 64 + sb) * 8 + tid) * 2 + 0] = s1;
    partials[(((size_t)layer * 64 + sb) * 8 + tid) * 2 + 1] = s2;
  }
  __syncthreads();
  if (tid == 0) {
    __threadfence();
    unsigned old = atomicAdd(&counters[layer], 1u);
    s_last = (old == 63u) ? 1 : 0;
  }
  __syncthreads();
  if (s_last) {                         // last WG of this layer: finalize LN stats
    __threadfence();
    if (tid < 8) {
      float s1 = 0.0f, s2 = 0.0f;
      for (int w = 0; w < 64; ++w) {
        s1 += partials[(((size_t)layer * 64 + w) * 8 + tid) * 2 + 0];
        s2 += partials[(((size_t)layer * 64 + w) * 8 + tid) * 2 + 1];
      }
      const float m   = s1 * (1.0f / S_DIM);
      const float var = s2 * (1.0f / S_DIM) - m * m;
      mu_buf[layer * 8 + tid]   = m;
      rstd_buf[layer * 8 + tid] = rsqrtf(var + 1e-5f);
    }
    __threadfence();
    if (tid == 0) atomicExch(&counters[layer], 0u);  // re-arm for next step/replay
  }
}

// ---------------- batched vocab projection ----------------
// out[b][t][n] = H_all[t*8+b] . Wout[n] + bout[n]
// grid: (V/16, 2048/256); WG = 8 waves, wave handles 32 rows x 16 cols.
__global__ __launch_bounds__(256) void out_gemm_kernel(
    const unsigned short* __restrict__ H_all,    // [2048][1024] bf16
    const unsigned short* __restrict__ Wout_bf,  // [32000][1024] bf16
    const float* __restrict__ bout,
    float* __restrict__ out) {
  const int tid  = threadIdx.x;
  const int wave = tid >> 5;
  const int lane = tid & 31;
  const int half = lane >> 4;
  const int l16  = lane & 15;

  const int n0 = blockIdx.x * 16;
  const int r0 = blockIdx.y * 256 + wave * 32;

  v8f acc0 = {}, acc1 = {};
  const unsigned short* wrow  = Wout_bf + (size_t)(n0 + l16) * S_DIM;
  const unsigned short* a0row = H_all + (size_t)(r0 + l16) * S_DIM;
  const unsigned short* a1row = a0row + (size_t)16 * S_DIM;

  #pragma unroll 4
  for (int kk = 0; kk < S_DIM; kk += 32) {
    v16bf bt = load_b_row(wrow, kk, half);
    v16bf a0 = load_a_row(a0row, kk, half);
    v16bf a1 = load_a_row(a1row, kk, half);
    acc0 = wmma_bf16(a0, bt, acc0);
    acc1 = wmma_bf16(a1, bt, acc1);
  }

  const float bias = bout[n0 + l16];
  #pragma unroll
  for (int v = 0; v < 8; ++v) {
    int r = r0 + v + half * 8;
    int bb = r & 7, tt = r >> 3;
    out[((size_t)bb * T_DIM + tt) * V_DIM + n0 + l16] = acc0[v] + bias;
    r += 16; bb = r & 7; tt = r >> 3;
    out[((size_t)bb * T_DIM + tt) * V_DIM + n0 + l16] = acc1[v] + bias;
  }
}

// ---------------- final h / c outputs ----------------
__global__ void final_copy_kernel(const float* __restrict__ h_f32,
                                  const float* __restrict__ c_pre,
                                  const float* __restrict__ mu_buf,
                                  const float* __restrict__ rstd_buf,
                                  const float* __restrict__ gamma,
                                  const float* __restrict__ beta,
                                  float* __restrict__ out_h,
                                  float* __restrict__ out_c) {
  int i = blockIdx.x * blockDim.x + threadIdx.x;   // 2*8*1024
  if (i >= 2 * B_DIM * S_DIM) return;
  int layer = i >> 13;
  int b = (i >> 10) & 7;
  int s = i & (S_DIM - 1);
  out_h[i] = h_f32[i];
  out_c[i] = (c_pre[i] - mu_buf[layer * 8 + b]) * rstd_buf[layer * 8 + b] *
                 gamma[layer * S_DIM + s] + beta[layer * S_DIM + s];
}

// ---------------- host launcher ----------------
extern "C" void kernel_launch(void* const* d_in, const int* in_sizes, int n_in,
                              void* d_out, int out_size, void* d_ws, size_t ws_size,
                              hipStream_t stream) {
  const int*   x     = (const int*)d_in[0];     // [B,T]
  const float* emb   = (const float*)d_in[1];   // [V,S]
  const float* Wx    = (const float*)d_in[2];   // [2,4096,1024]
  const float* Wh    = (const float*)d_in[3];
  const float* gamma = (const float*)d_in[4];   // [2,1024]
  const float* beta  = (const float*)d_in[5];
  const float* Wout  = (const float*)d_in[6];   // [V,S]
  const float* bout  = (const float*)d_in[7];   // [V]
  float* out = (float*)d_out;

  char* ws = (char*)d_ws;
  size_t off = 0;
  auto take = [&](size_t bytes) { size_t cur = off; off += (bytes + 255) & ~(size_t)255; return cur; };

  const size_t nWx   = 2ull * 4 * S_DIM * S_DIM;        // 8,388,608 elems
  const size_t nWout = (size_t)V_DIM * S_DIM;           // 32,768,000 elems
  const size_t nSeqP = (size_t)T_DIM * ROW16;           // padded input: 4,194,304 elems
  const size_t nHall = (size_t)T_DIM * B_DIM * S_DIM;   // 2,097,152 elems

  size_t off_Wx   = take(nWx * 2);
  size_t off_Wh   = take(nWx * 2);
  size_t off_Wo   = take(nWout * 2);
  size_t off_inp  = take(nSeqP * 2);
  size_t off_Hall = take(nHall * 2);
  size_t zero_begin = off;
  size_t off_h0   = take(2ull * ROW16 * 2);
  size_t off_h1   = take((size_t)ROW16 * 2);
  size_t off_hf   = take(2ull * B_DIM * S_DIM * 4);
  size_t off_cpre = take(2ull * B_DIM * S_DIM * 4);
  size_t off_mu   = take(2 * 8 * 4);
  size_t off_rstd = take(2 * 8 * 4);
  size_t off_part = take(2ull * 64 * 8 * 2 * 4);
  size_t off_cnt  = take(2 * 4);
  size_t zero_end = off;

  unsigned short* Wx_bf   = (unsigned short*)(ws + off_Wx);
  unsigned short* Wh_bf   = (unsigned short*)(ws + off_Wh);
  unsigned short* Wout_bf = (unsigned short*)(ws + off_Wo);
  unsigned short* inp_seq = (unsigned short*)(ws + off_inp);
  unsigned short* H_all   = (unsigned short*)(ws + off_Hall);
  unsigned short* h0_buf  = (unsigned short*)(ws + off_h0);
  unsigned short* h1_buf  = (unsigned short*)(ws + off_h1);
  float*        h_f32     = (float*)(ws + off_hf);
  float*        c_pre     = (float*)(ws + off_cpre);
  float*        mu_buf    = (float*)(ws + off_mu);
  float*        rstd_buf  = (float*)(ws + off_rstd);
  float*        partials  = (float*)(ws + off_part);
  unsigned int* counters  = (unsigned int*)(ws + off_cnt);

  // 1) zero recurrent state (incl. padded h rows; ws is poisoned, not re-poisoned)
  int nwords = (int)((zero_end - zero_begin) / 4);
  zero_ws_kernel<<<(nwords + 255) / 256, 256, 0, stream>>>(
      (unsigned int*)(ws + zero_begin), nwords);

  // 2) one-time bf16 weight conversion (per call; trivial vs. the scan)
  convert_bf16_kernel<<<(int)((nWx + 255) / 256), 256, 0, stream>>>(Wx, Wx_bf, (int)nWx);
  convert_bf16_kernel<<<(int)((nWx + 255) / 256), 256, 0, stream>>>(Wh, Wh_bf, (int)nWx);
  convert_bf16_kernel<<<(int)((nWout + 255) / 256), 256, 0, stream>>>(Wout, Wout_bf, (int)nWout);

  // 3) embedding gather for every timestep (writes zero padding rows too)
  gather_emb_kernel<<<(int)(nSeqP / 256), 256, 0, stream>>>(x, emb, inp_seq);

  // 4) sequential scan: prologue L0(0); fused kernels {L1(t), L0(t+1)}; epilogue L1(T-1)
  lstm_step_kernel<<<64, 256, 0, stream>>>(Wx_bf, Wh_bf, inp_seq, h0_buf, h1_buf,
      h_f32, c_pre, mu_buf, rstd_buf, partials, counters, H_all, gamma, beta,
      /*t1=*/-1, /*t0=*/0, /*n1=*/0);
  for (int t = 0; t < T_DIM - 1; ++t) {
    lstm_step_kernel<<<128, 256, 0, stream>>>(Wx_bf, Wh_bf, inp_seq, h0_buf, h1_buf,
        h_f32, c_pre, mu_buf, rstd_buf, partials, counters, H_all, gamma, beta,
        /*t1=*/t, /*t0=*/t + 1, /*n1=*/64);
  }
  lstm_step_kernel<<<64, 256, 0, stream>>>(Wx_bf, Wh_bf, inp_seq, h0_buf, h1_buf,
      h_f32, c_pre, mu_buf, rstd_buf, partials, counters, H_all, gamma, beta,
      /*t1=*/T_DIM - 1, /*t0=*/-1, /*n1=*/64);

  // 5) single batched vocab projection for all (b,t)
  out_gemm_kernel<<<dim3(V_DIM / 16, (T_DIM * B_DIM) / 256), 256, 0, stream>>>(
      H_all, Wout_bf, bout, out);

  // 6) final h and normalized c
  float* out_h = out + (size_t)B_DIM * T_DIM * V_DIM;
  float* out_c = out_h + 2ull * B_DIM * S_DIM;
  final_copy_kernel<<<(2 * B_DIM * S_DIM + 255) / 256, 256, 0, stream>>>(
      h_f32, c_pre, mu_buf, rstd_buf, gamma, beta, out_h, out_c);
}
